// SpanAsteModel_35553739276883
// MI455X (gfx1250) — compile-verified, matrix operand order checked by reading, not run on hardware
//
#include <hip/hip_runtime.h>
#include <hip/hip_bf16.h>

typedef __bf16 bf16_t;
typedef __attribute__((ext_vector_type(16))) __bf16 v16bf;
typedef __attribute__((ext_vector_type(8)))  float  v8f;

#define NPAD   160      // padded hidden width (150 -> 160, 10 WMMA n-tiles)
#define NTILES 10       // NPAD / 16
#define SEQL   256
#define BATCH  2
#define HDIM   768
#define NSPANS 2020
#define SPAD   2048
#define NZ     128
#define NPAIR  (NZ*NZ)  // 16384

// ---------- small device helpers ----------

__device__ __forceinline__ int bucket_of(int width) {
  const int bins[14] = {0,1,2,3,4,5,7,8,15,16,31,32,63,64};
  int ix = 0;
#pragma unroll
  for (int i = 1; i < 14; ++i) ix = (width >= bins[i]) ? i : ix;
  return ix;
}

// spans enumerated window w=1..8, starts 0..L-w  (same order as reference)
__device__ __forceinline__ void span_decode(int s, int& st, int& en, int& w) {
  int off = 0; st = 0; en = 0; w = 1;
#pragma unroll
  for (int wi = 1; wi <= 8; ++wi) {
    int cnt = SEQL - wi + 1;
    if (s >= off && s < off + cnt) { w = wi; st = s - off; en = st + wi - 1; }
    off += cnt;
  }
}

// ---------- conversion / packing kernels ----------

__global__ __launch_bounds__(256)
void cvt_x_kernel(const float* __restrict__ x, bf16_t* __restrict__ xbf, int n) {
  int i = blockIdx.x * 256 + threadIdx.x;
  if (i < n) xbf[i] = (bf16_t)x[i];
}

// Pack fp32 [Ksrc x 150] into WMMA-B fragment-major bf16 layout:
//   dst[((ks*NTILES + nt)*32 + lane)*16 + e] = B[ks*32 + kk][nt*16 + lidx]
//   kk = (e>>3)*16 + (lane>>4)*8 + (e&7),  lidx = lane&15. Zero padded.
__global__ __launch_bounds__(256)
void pack_w_kernel(const float* __restrict__ src, int Ksrc, int Kpad,
                   bf16_t* __restrict__ dst) {
  int i = blockIdx.x * 256 + threadIdx.x;
  int total = (Kpad / 32) * NTILES * 32 * 16;   // == Kpad * NPAD
  if (i >= total) return;
  int e    = i & 15;
  int lane = (i >> 4) & 31;
  int nt   = (i >> 9) % NTILES;
  int ks   = i / (16 * 32 * NTILES);
  int half = lane >> 4, lidx = lane & 15;
  int kk = (e >> 3) * 16 + half * 8 + (e & 7);
  int r = ks * 32 + kk;
  int c = nt * 16 + lidx;
  float v = (r < Ksrc && c < 150) ? src[(size_t)r * 150 + c] : 0.f;
  dst[i] = (bf16_t)v;
}

// Precompute embedding@W1-slice tables and zero-padded biases.
__global__ __launch_bounds__(256)
void tables_kernel(const float* __restrict__ w_width, const float* __restrict__ span_w1,
                   const float* __restrict__ pair_w1, const float* __restrict__ w_dist,
                   const float* __restrict__ span_b1, const float* __restrict__ span_b2,
                   const float* __restrict__ pair_b1, const float* __restrict__ pair_b2,
                   float* __restrict__ WEs, float* __restrict__ WEt, float* __restrict__ WEo,
                   float* __restrict__ DW,
                   float* __restrict__ b1s, float* __restrict__ b2s,
                   float* __restrict__ b1p, float* __restrict__ b2p) {
  int tid = threadIdx.x;
  for (int i = tid; i < 8 * NPAD; i += 256) {
    int w = i / NPAD + 1, j = i % NPAD;
    int bk = bucket_of(w);
    float vs = 0.f, vt = 0.f, vo = 0.f;
    if (j < 150) {
      for (int d = 0; d < 20; ++d) {
        float e = w_width[bk * 20 + d];
        vs += e * span_w1[(size_t)(1536 + d) * 150 + j];
        vt += e * pair_w1[(size_t)(1536 + d) * 150 + j];
        vo += e * pair_w1[(size_t)(3092 + d) * 150 + j];
      }
    }
    WEs[i] = vs; WEt[i] = vt; WEo[i] = vo;
  }
  for (int i = tid; i < 14 * NPAD; i += 256) {
    int m = i / NPAD, j = i % NPAD;
    float v = 0.f;
    if (j < 150)
      for (int d = 0; d < 128; ++d)
        v += w_dist[m * 128 + d] * pair_w1[(size_t)(3112 + d) * 150 + j];
    DW[i] = v;
  }
  for (int j = tid; j < NPAD; j += 256) {
    b1s[j] = (j < 150) ? span_b1[j] : 0.f;
    b2s[j] = (j < 150) ? span_b2[j] : 0.f;
    b1p[j] = (j < 150) ? pair_b1[j] : 0.f;
    b2p[j] = (j < 150) ? pair_b2[j] : 0.f;
  }
}

// ---------- generic bf16 WMMA GEMM: C[MxNPAD] = A[MxK] * B[KxNPAD] ----------
// B is pre-swizzled fragment-major (pack_w_kernel): each lane's fragment is one
// aligned 32-byte global load. A staged through LDS (row-major, 2x ds_load_b128
// per fragment). 128 threads = 4 waves; block computes 64 rows x all 10 n-tiles.

__global__ __launch_bounds__(128)
void gemm_bf16_wmma(const bf16_t* __restrict__ A, int lda,
                    const v16bf* __restrict__ Bswz,
                    const float* __restrict__ bias,
                    bf16_t* __restrict__ outB, float* __restrict__ outF,
                    int M, int K, int relu) {
  __shared__ bf16_t sA[64 * 32];
  int tid  = threadIdx.x;
  int wave = tid >> 5;
  int lane = tid & 31;
  int half = lane >> 4;
  int lidx = lane & 15;
  int mblk = blockIdx.x * 64;

  v8f acc[NTILES] = {};

  for (int k0 = 0, ks = 0; k0 < K; k0 += 32, ++ks) {
    // stage A tile 64x32 (2 threads per row, 16 contiguous bf16 each)
    {
      int r = tid >> 1, c = (tid & 1) * 16;
      int gr = mblk + r;
      if (gr < M) {
        const bf16_t* src = A + (size_t)gr * lda + k0 + c;
#pragma unroll
        for (int e = 0; e < 16; ++e) sA[r * 32 + c + e] = src[e];
      } else {
#pragma unroll
        for (int e = 0; e < 16; ++e) sA[r * 32 + c + e] = (bf16_t)0.f;
      }
    }
    __syncthreads();

    // A fragment: lanes 0-15 hold K 0..7,16..23; lanes 16-31 hold K 8..15,24..31
    v16bf afrag;
#pragma unroll
    for (int e = 0; e < 16; ++e) {
      int g = e >> 3, w = e & 7;
      int kk = g * 16 + half * 8 + w;
      afrag[e] = sA[(wave * 16 + lidx) * 32 + kk];
    }

    const v16bf* bp = Bswz + ((size_t)ks * NTILES) * 32 + lane;
#pragma unroll
    for (int nt = 0; nt < NTILES; ++nt) {
      v16bf bfrag = bp[nt * 32];           // one aligned 32B load per lane
      acc[nt] = __builtin_amdgcn_wmma_f32_16x16x32_bf16(
          false, afrag, false, bfrag, (short)0, acc[nt], false, false);
    }
    __syncthreads();
  }

  // C layout: VGPR r -> row r (lanes 0-15) / r+8 (lanes 16-31); col = lane%16
#pragma unroll
  for (int nt = 0; nt < NTILES; ++nt) {
#pragma unroll
    for (int r = 0; r < 8; ++r) {
      int m = mblk + wave * 16 + half * 8 + r;
      int n = nt * 16 + lidx;
      if (m < M) {
        float v = acc[nt][r];
        if (bias) v += bias[n];
        if (relu) v = fmaxf(v, 0.f);
        if (outF) outF[(size_t)m * NPAD + n] = v;
        if (outB) outB[(size_t)m * NPAD + n] = (bf16_t)v;
      }
    }
  }
}

// ---------- span stage ----------

__global__ __launch_bounds__(NPAD)
void span_h1_kernel(const float* __restrict__ XA, const float* __restrict__ XB,
                    const float* __restrict__ WEs, const float* __restrict__ b1s,
                    bf16_t* __restrict__ H1s) {
  int row = blockIdx.x;            // b*SPAD + s
  int b = row >> 11, s = row & (SPAD - 1);
  int j = threadIdx.x;
  float v = 0.f;
  if (s < NSPANS) {
    int st, en, w; span_decode(s, st, en, w);
    v = XA[((size_t)(b * SEQL + st)) * NPAD + j] +
        XB[((size_t)(b * SEQL + en)) * NPAD + j] +
        WEs[(w - 1) * NPAD + j] + b1s[j];
    v = fmaxf(v, 0.f);
  }
  H1s[(size_t)row * NPAD + j] = (bf16_t)v;
}

__global__ __launch_bounds__(256)
void span_out_kernel(const bf16_t* __restrict__ H2s, const float* __restrict__ w3,
                     const float* __restrict__ b3, float* __restrict__ out) {
  int idx = blockIdx.x * 256 + threadIdx.x;
  if (idx >= BATCH * NSPANS) return;
  int b = idx / NSPANS, s = idx % NSPANS;
  const bf16_t* h = H2s + ((size_t)b * SPAD + s) * NPAD;
  float l0 = b3[0], l1 = b3[1], l2 = b3[2];
  for (int j = 0; j < 150; ++j) {
    float hv = (float)h[j];
    l0 += hv * w3[j * 3 + 0];
    l1 += hv * w3[j * 3 + 1];
    l2 += hv * w3[j * 3 + 2];
  }
  float m = fmaxf(l0, fmaxf(l1, l2));
  float e0 = __expf(l0 - m), e1 = __expf(l1 - m), e2 = __expf(l2 - m);
  float inv = 1.f / (e0 + e1 + e2);
  out[(size_t)idx * 3 + 0] = e0 * inv;
  out[(size_t)idx * 3 + 1] = e1 * inv;
  out[(size_t)idx * 3 + 2] = e2 * inv;
}

// ---------- exact top-k via full bitonic sort (desc value, asc index on ties) ----------

__global__ __launch_bounds__(256)
void topk_kernel(const float* __restrict__ spans_prob,
                 int* __restrict__ tsel, int* __restrict__ osel) {
  __shared__ float sv[SPAD];
  __shared__ int   si[SPAD];
  int b = blockIdx.x >> 1;
  int label = (blockIdx.x & 1) + 1;     // 1 = ASPECT, 2 = OPINION
  int tid = threadIdx.x;
  for (int i = tid; i < SPAD; i += 256) {
    sv[i] = (i < NSPANS) ? spans_prob[((size_t)b * NSPANS + i) * 3 + label]
                         : -__builtin_inff();
    si[i] = i;
  }
  __syncthreads();
  for (unsigned k = 2; k <= SPAD; k <<= 1) {
    for (unsigned j = k >> 1; j > 0; j >>= 1) {
      for (unsigned i = tid; i < SPAD; i += 256) {
        unsigned ixj = i ^ j;
        if (ixj > i) {
          float v0 = sv[i], v1 = sv[ixj];
          int d0 = si[i], d1 = si[ixj];
          bool before = (v0 > v1) || (v0 == v1 && d0 < d1);
          bool descRegion = ((i & k) == 0);
          if (before != descRegion) {
            sv[i] = v1; sv[ixj] = v0; si[i] = d1; si[ixj] = d0;
          }
        }
      }
      __syncthreads();
    }
  }
  int* sel = (label == 1) ? tsel : osel;
  for (int i = tid; i < NZ; i += 256) sel[b * NZ + i] = si[i];
}

// ---------- pair stage ----------

__global__ __launch_bounds__(256)
void gather_kernel(const bf16_t* __restrict__ xbf,
                   const int* __restrict__ tsel, const int* __restrict__ osel,
                   bf16_t* __restrict__ Gt, bf16_t* __restrict__ Go,
                   int* __restrict__ meta) {   // meta: 6 arrays of [BATCH*NZ]
  int blk = blockIdx.x;           // b*NZ + i
  int b = blk >> 7;
  int tid = threadIdx.x;
  int sT = tsel[blk], sO = osel[blk];
  int ts, te, tw, os_, oe, ow;
  span_decode(sT, ts, te, tw);
  span_decode(sO, os_, oe, ow);
  if (tid == 0) {
    const int N = BATCH * NZ;
    meta[0 * N + blk] = ts;  meta[1 * N + blk] = te;  meta[2 * N + blk] = tw;
    meta[3 * N + blk] = os_; meta[4 * N + blk] = oe;  meta[5 * N + blk] = ow;
  }
  const bf16_t* xb = xbf + (size_t)b * SEQL * HDIM;
  size_t row = (size_t)blk * (2 * HDIM);
  for (int c = tid; c < HDIM; c += 256) {
    Gt[row + c]        = xb[(size_t)ts * HDIM + c];
    Gt[row + HDIM + c] = xb[(size_t)te * HDIM + c];
    Go[row + c]        = xb[(size_t)os_ * HDIM + c];
    Go[row + HDIM + c] = xb[(size_t)oe * HDIM + c];
  }
}

__global__ __launch_bounds__(NPAD)
void pair_h1_kernel(const float* __restrict__ Ttf, const float* __restrict__ Tof,
                    const float* __restrict__ WEt, const float* __restrict__ WEo,
                    const float* __restrict__ DW, const float* __restrict__ b1p,
                    const int* __restrict__ meta, bf16_t* __restrict__ H1p) {
  int r = blockIdx.x;                      // b*NPAIR + i*NZ + j
  int b = r >> 14, i = (r >> 7) & 127, j = r & 127;
  int t = threadIdx.x;
  const int N = BATCH * NZ;
  int ti = b * NZ + i, oj = b * NZ + j;
  int ts  = meta[0 * N + ti], te = meta[1 * N + ti], tw = meta[2 * N + ti];
  int os_ = meta[3 * N + oj], oe = meta[4 * N + oj], ow = meta[5 * N + oj];
  int d1 = te - os_; if (d1 < 0) d1 = -d1;
  int d2 = ts - oe;  if (d2 < 0) d2 = -d2;
  int md = d1 < d2 ? d1 : d2;
  int bk = bucket_of(md);
  float v = Ttf[(size_t)ti * NPAD + t] + Tof[(size_t)oj * NPAD + t] +
            DW[bk * NPAD + t] + WEt[(tw - 1) * NPAD + t] +
            WEo[(ow - 1) * NPAD + t] + b1p[t];
  v = fmaxf(v, 0.f);
  H1p[(size_t)r * NPAD + t] = (bf16_t)v;
}

__global__ __launch_bounds__(256)
void pair_out_kernel(const bf16_t* __restrict__ H2p, const float* __restrict__ w3,
                     const float* __restrict__ b3, float* __restrict__ out) {
  int r = blockIdx.x * 256 + threadIdx.x;
  if (r >= BATCH * NPAIR) return;
  const bf16_t* h = H2p + (size_t)r * NPAD;
  float l[4] = {b3[0], b3[1], b3[2], b3[3]};
  for (int j = 0; j < 150; ++j) {
    float hv = (float)h[j];
    l[0] += hv * w3[j * 4 + 0];
    l[1] += hv * w3[j * 4 + 1];
    l[2] += hv * w3[j * 4 + 2];
    l[3] += hv * w3[j * 4 + 3];
  }
  float m = fmaxf(fmaxf(l[0], l[1]), fmaxf(l[2], l[3]));
  float e0 = __expf(l[0] - m), e1 = __expf(l[1] - m),
        e2 = __expf(l[2] - m), e3 = __expf(l[3] - m);
  float inv = 1.f / (e0 + e1 + e2 + e3);
  out[(size_t)r * 4 + 0] = e0 * inv;
  out[(size_t)r * 4 + 1] = e1 * inv;
  out[(size_t)r * 4 + 2] = e2 * inv;
  out[(size_t)r * 4 + 3] = e3 * inv;
}

// ---------- host launcher ----------

static inline size_t alignup256(size_t v) { return (v + 255) & ~(size_t)255; }

extern "C" void kernel_launch(void* const* d_in, const int* in_sizes, int n_in,
                              void* d_out, int out_size, void* d_ws, size_t ws_size,
                              hipStream_t stream) {
  (void)in_sizes; (void)n_in; (void)out_size; (void)ws_size;
  const float* x       = (const float*)d_in[0];
  const float* w_width = (const float*)d_in[1];
  const float* span_w1 = (const float*)d_in[2];
  const float* span_b1 = (const float*)d_in[3];
  const float* span_w2 = (const float*)d_in[4];
  const float* span_b2 = (const float*)d_in[5];
  const float* span_w3 = (const float*)d_in[6];
  const float* span_b3 = (const float*)d_in[7];
  const float* w_dist  = (const float*)d_in[8];
  const float* pair_w1 = (const float*)d_in[9];
  const float* pair_b1 = (const float*)d_in[10];
  const float* pair_w2 = (const float*)d_in[11];
  const float* pair_b2 = (const float*)d_in[12];
  const float* pair_w3 = (const float*)d_in[13];
  const float* pair_b3 = (const float*)d_in[14];
  float* out = (float*)d_out;

  char* base = (char*)d_ws;
  size_t off = 0;
  auto alloc = [&](size_t bytes) -> void* {
    void* r = base + off;
    off = alignup256(off + bytes);
    return r;
  };

  bf16_t* xbf = (bf16_t*)alloc((size_t)BATCH * SEQL * HDIM * 2);
  bf16_t* W1a = (bf16_t*)alloc((size_t)HDIM * NPAD * 2);
  bf16_t* W1b = (bf16_t*)alloc((size_t)HDIM * NPAD * 2);
  bf16_t* W2s = (bf16_t*)alloc((size_t)NPAD * NPAD * 2);
  bf16_t* P1t = (bf16_t*)alloc((size_t)(2 * HDIM) * NPAD * 2);
  bf16_t* P1o = (bf16_t*)alloc((size_t)(2 * HDIM) * NPAD * 2);
  bf16_t* W2p = (bf16_t*)alloc((size_t)NPAD * NPAD * 2);
  float* WEs = (float*)alloc(8 * NPAD * 4);
  float* WEt = (float*)alloc(8 * NPAD * 4);
  float* WEo = (float*)alloc(8 * NPAD * 4);
  float* DW  = (float*)alloc(14 * NPAD * 4);
  float* b1s = (float*)alloc(NPAD * 4);
  float* b2s = (float*)alloc(NPAD * 4);
  float* b1p = (float*)alloc(NPAD * 4);
  float* b2p = (float*)alloc(NPAD * 4);
  float* XA  = (float*)alloc((size_t)BATCH * SEQL * NPAD * 4);
  float* XB  = (float*)alloc((size_t)BATCH * SEQL * NPAD * 4);
  bf16_t* H1s = (bf16_t*)alloc((size_t)BATCH * SPAD * NPAD * 2);
  bf16_t* H2s = (bf16_t*)alloc((size_t)BATCH * SPAD * NPAD * 2);
  int* tsel = (int*)alloc(BATCH * NZ * 4);
  int* osel = (int*)alloc(BATCH * NZ * 4);
  int* meta = (int*)alloc(6 * BATCH * NZ * 4);
  bf16_t* Gt  = (bf16_t*)alloc((size_t)BATCH * NZ * 2 * HDIM * 2);
  bf16_t* Go  = (bf16_t*)alloc((size_t)BATCH * NZ * 2 * HDIM * 2);
  float* Ttf = (float*)alloc((size_t)BATCH * NZ * NPAD * 4);
  float* Tof = (float*)alloc((size_t)BATCH * NZ * NPAD * 4);
  bf16_t* H1p = (bf16_t*)alloc((size_t)BATCH * NPAIR * NPAD * 2);
  bf16_t* H2p = (bf16_t*)alloc((size_t)BATCH * NPAIR * NPAD * 2);

  float* spans_prob = out;                               // [2,2020,3]
  float* cand_prob  = out + (size_t)BATCH * NSPANS * 3;  // [2,16384,4]

  // 1) x -> bf16
  {
    int n = BATCH * SEQL * HDIM;
    cvt_x_kernel<<<(n + 255) / 256, 256, 0, stream>>>(x, xbf, n);
  }
  // 2) weight packing into WMMA fragment-major bf16 (zero padded)
  pack_w_kernel<<<(HDIM * NPAD + 255) / 256, 256, 0, stream>>>(span_w1, HDIM, HDIM, W1a);
  pack_w_kernel<<<(HDIM * NPAD + 255) / 256, 256, 0, stream>>>(span_w1 + (size_t)HDIM * 150, HDIM, HDIM, W1b);
  pack_w_kernel<<<(NPAD * NPAD + 255) / 256, 256, 0, stream>>>(span_w2, 150, NPAD, W2s);
  pack_w_kernel<<<(2 * HDIM * NPAD + 255) / 256, 256, 0, stream>>>(pair_w1, 2 * HDIM, 2 * HDIM, P1t);
  pack_w_kernel<<<(2 * HDIM * NPAD + 255) / 256, 256, 0, stream>>>(pair_w1 + (size_t)1556 * 150, 2 * HDIM, 2 * HDIM, P1o);
  pack_w_kernel<<<(NPAD * NPAD + 255) / 256, 256, 0, stream>>>(pair_w2, 150, NPAD, W2p);
  // 3) embedding/bias tables
  tables_kernel<<<1, 256, 0, stream>>>(w_width, span_w1, pair_w1, w_dist,
                                       span_b1, span_b2, pair_b1, pair_b2,
                                       WEs, WEt, WEo, DW, b1s, b2s, b1p, b2p);
  // 4) span layer-1 pieces: XA/XB = x @ W1a/W1b   (M=512, K=768)
  gemm_bf16_wmma<<<(BATCH * SEQL + 63) / 64, 128, 0, stream>>>(
      xbf, HDIM, (const v16bf*)W1a, nullptr, nullptr, XA, BATCH * SEQL, HDIM, 0);
  gemm_bf16_wmma<<<(BATCH * SEQL + 63) / 64, 128, 0, stream>>>(
      xbf, HDIM, (const v16bf*)W1b, nullptr, nullptr, XB, BATCH * SEQL, HDIM, 0);
  // 5) fused span h1 = relu(XA[start]+XB[end]+WE[w]+b1) -> bf16 [B*SPAD, NPAD]
  span_h1_kernel<<<BATCH * SPAD, NPAD, 0, stream>>>(XA, XB, WEs, b1s, H1s);
  // 6) span layer-2: H2s = relu(H1s @ W2s + b2)  (M=4096, K=160)
  gemm_bf16_wmma<<<(BATCH * SPAD + 63) / 64, 128, 0, stream>>>(
      H1s, NPAD, (const v16bf*)W2s, b2s, H2s, nullptr, BATCH * SPAD, NPAD, 1);
  // 7) span layer-3 + softmax -> spans_prob
  span_out_kernel<<<(BATCH * NSPANS + 255) / 256, 256, 0, stream>>>(
      H2s, span_w3, span_b3, spans_prob);
  // 8) exact top-128 per (batch, label)
  topk_kernel<<<BATCH * 2, 256, 0, stream>>>(spans_prob, tsel, osel);
  // 9) gather selected span token rows (concat start|end -> 1536 wide)
  gather_kernel<<<BATCH * NZ, 256, 0, stream>>>(xbf, tsel, osel, Gt, Go, meta);
  // 10) pair layer-1 pieces: Tt/To = G @ P1  (M=256, K=1536)
  gemm_bf16_wmma<<<(BATCH * NZ + 63) / 64, 128, 0, stream>>>(
      Gt, 2 * HDIM, (const v16bf*)P1t, nullptr, nullptr, Ttf, BATCH * NZ, 2 * HDIM, 0);
  gemm_bf16_wmma<<<(BATCH * NZ + 63) / 64, 128, 0, stream>>>(
      Go, 2 * HDIM, (const v16bf*)P1o, nullptr, nullptr, Tof, BATCH * NZ, 2 * HDIM, 0);
  // 11) fused pair h1 = relu(Tt[i]+To[j]+DW[bucket]+WEt+WEo+b1)
  pair_h1_kernel<<<BATCH * NPAIR, NPAD, 0, stream>>>(Ttf, Tof, WEt, WEo, DW, b1p, meta, H1p);
  // 12) pair layer-2: H2p = relu(H1p @ W2p + b2)  (M=32768, K=160)
  gemm_bf16_wmma<<<(BATCH * NPAIR + 63) / 64, 128, 0, stream>>>(
      H1p, NPAD, (const v16bf*)W2p, b2p, H2p, nullptr, BATCH * NPAIR, NPAD, 1);
  // 13) pair layer-3 + softmax -> cand_prob
  pair_out_kernel<<<(BATCH * NPAIR + 255) / 256, 256, 0, stream>>>(
      H2p, pair_w3, pair_b3, cand_prob);
}
